// TIFM_84112639524881
// MI455X (gfx1250) — matrix operand order repaired and checked by reading, not compiled, not run on hardware
//
#include <hip/hip_runtime.h>
#include <cstdint>
#include <cmath>

// ---------------------------------------------------------------------------
// TIFM pipeline for MI455X (gfx1250, wave32, WMMA + TDM).
// All channel contractions (3x3 convs, 1x1 deform-GEMM) run on
// v_wmma_f32_16x16x32_f16 via implicit GEMM; gathers stay on VALU.
// Block tile: 64 Cout (4 WMMA M-tiles) x 64 pixels; one im2col B fragment
// is reused by 4 WMMAs. A-tiles are DMA'd into LDS by the Tensor Data Mover
// (tensor_load_to_lds + s_wait_tensorcnt) when the builtin is available,
// else staged manually.
// ---------------------------------------------------------------------------

typedef __attribute__((ext_vector_type(16))) _Float16 v16h;
typedef __attribute__((ext_vector_type(8)))  float    v8f;
typedef __attribute__((ext_vector_type(4)))  unsigned int u32x4;
typedef __attribute__((ext_vector_type(8)))  int      i32x8;
typedef __attribute__((ext_vector_type(4)))  int      i32x4;

#if defined(__has_builtin)
#if __has_builtin(__builtin_amdgcn_tensor_load_to_lds) && \
    __has_builtin(__builtin_amdgcn_s_wait_tensorcnt)
#define TIFM_USE_TDM 1
#endif
#endif

#define NF   64
#define TT   7
#define HH   192
#define WW   192
#define NPIX (HH * WW)
#define MT_TILES 4

static inline int ceil_div_h(int a, int b) { return (a + b - 1) / b; }

// ---------------------------------------------------------------------------
// Weight packing: f32 [Cout][Ktot] (OIHW flattened) -> f16 WMMA A-fragments.
// Layout per 16x32 tile, per ISA 7.12.2 (16-bit A 16x32):
//   lane = (m % 16) + 16*halfSel ; half j (0..15): K = 32*kt + 16*(j/8) + 8*halfSel + (j%8)
// Stored as Ap[((mtile*kt + ktile)*32 + lane)*16 + j].  mt is padded to a
// multiple of MT_TILES (zero-filled) so the conv kernel can blind-stage.
// ---------------------------------------------------------------------------
__global__ void pack_w_kernel(const float* __restrict__ w, _Float16* __restrict__ Ap,
                              int Cout, int Ktot, int mt, int kt) {
  int idx = blockIdx.x * 256 + threadIdx.x;
  int total = mt * kt * 512;
  if (idx >= total) return;
  int j    = idx & 15;
  int lane = (idx >> 4) & 31;
  int tile = idx >> 9;
  int ktile = tile % kt;
  int mtile = tile / kt;
  int m = mtile * 16 + (lane & 15);
  int h = lane >> 4;
  int k = ktile * 32 + 16 * (j >> 3) + 8 * h + (j & 7);
  float v = 0.f;
  if (m < Cout && k < Ktot) v = w[(size_t)m * Ktot + k];
  Ap[idx] = (_Float16)v;
}

// ---------------------------------------------------------------------------
// Implicit-GEMM conv, KS compile-time (1 or 3), pad=KS/2, stride 1 or 2.
// 128 threads = 4 waves; each wave owns 16 output pixels, block owns
// MT*16 output channels. Per K-tile: MT A-tiles DMA'd to LDS (TDM), one B
// fragment im2col'd per wave, MT WMMAs issued against it.
// Epilogue: +bias, act(0:none,1:relu,2:lrelu0.1).
// ---------------------------------------------------------------------------
template <int KS, int MT>
__global__ void __launch_bounds__(128)
conv_wmma_kernel(const float* __restrict__ x, const _Float16* __restrict__ Ap,
                 const float* __restrict__ bias, float* __restrict__ y,
                 int Cin, int H, int W, int Cout, int Ho, int Wo,
                 int stride, int ktiles, int act) {
  __shared__ __align__(32) _Float16 As[MT * 512];
  constexpr int KSQ = KS * KS;

  const int npix = Ho * Wo;
  const int lane = threadIdx.x & 31;
  const int wave = threadIdx.x >> 5;
  const int col  = lane & 15;
  const int half = lane >> 4;
  const int pix  = blockIdx.y * 64 + wave * 16 + col;
  const bool pv  = (pix < npix);
  const int oy   = pv ? (pix / Wo) : 0;
  const int ox   = pv ? (pix % Wo) : 0;
  const int Ktot = Cin * KSQ;
  const int mbase = blockIdx.x * MT;

  x += (size_t)blockIdx.z * Cin * H * W;
  y += (size_t)blockIdx.z * Cout * npix;

  v8f acc[MT];
#pragma unroll
  for (int i = 0; i < MT; ++i) acc[i] = (v8f){0.f, 0.f, 0.f, 0.f, 0.f, 0.f, 0.f, 0.f};

  for (int kt = 0; kt < ktiles; ++kt) {
    __syncthreads();  // previous LDS reads complete
#if defined(TIFM_USE_TDM) && defined(__HIP_DEVICE_COMPILE__)
    // Tensor Data Mover: 2D tile, MT rows of 256 dwords, row stride
    // ktiles*256 dwords, into LDS at As. Issued by wave 0 only.
    if (threadIdx.x < 32) {
      uint64_t gbase = (uint64_t)(uintptr_t)(Ap + ((size_t)mbase * ktiles + kt) * 512);
      uint32_t ldsb  = (uint32_t)(uintptr_t)(&As[0]);
      u32x4 g0 = {
          1u,                                   // count=1, user descriptor
          ldsb,                                 // lds_addr (bytes)
          (unsigned int)gbase,                  // global_addr[31:0]
          (((unsigned int)(gbase >> 32)) & 0x1ffffffu) | (2u << 30)  // addr hi, type=2
      };
      i32x8 g1 = {
          (int)(2u << 16),                      // data_size = 4 bytes
          (int)(256u << 16),                    // tensor_dim0[15:0]=256
          (int)((unsigned)MT << 16),            // dim0 hi=0 | tensor_dim1=MT
          (int)(256u << 16),                    // dim1 hi=0 | tile_dim0=256
          MT,                                   // tile_dim1=MT, tile_dim2=0
          ktiles * 256,                         // tensor_dim0_stride[31:0]
          0, 0
      };
      i32x4 g2 = {0, 0, 0, 0};
      i32x4 g3 = {0, 0, 0, 0};
#if __clang_major__ >= 23
      i32x8 g4 = {0, 0, 0, 0, 0, 0, 0, 0};
      __builtin_amdgcn_tensor_load_to_lds(g0, g1, g2, g3, g4, 0);
#else
      __builtin_amdgcn_tensor_load_to_lds(g0, g1, g2, g3, 0);
#endif
      __builtin_amdgcn_s_wait_tensorcnt(0);
    }
    __syncthreads();
#else
    // fallback: manual staging, MT*64 uint4 total, 128 threads
#pragma unroll
    for (int i = 0; i < (MT * 64) / 128; ++i) {
      int idx = threadIdx.x + i * 128;
      int mi  = idx >> 6;
      int wi  = idx & 63;
      const uint4* src = (const uint4*)(Ap + ((size_t)(mbase + mi) * ktiles + kt) * 512);
      ((uint4*)As)[mi * 64 + wi] = src[wi];
    }
    __syncthreads();
#endif
    if (kt + 1 < ktiles)  // global_prefetch_b8 of next A group
      __builtin_prefetch((const void*)(Ap + ((size_t)mbase * ktiles + kt + 1) * 512), 0, 1);

    // build one B fragment per wave (im2col, f32 -> f16)
    const int kbase = kt * 32 + half * 8;
    v16h b;
#pragma unroll
    for (int j = 0; j < 16; ++j) {
      int k = kbase + 16 * (j >> 3) + (j & 7);
      float v = 0.f;
      if (pv && k < Ktot) {
        int ci, iy, ix;
        if (KS == 1) {
          ci = k; iy = oy; ix = ox;
        } else {
          ci = k / KSQ;                 // constant divisor -> mul/shift
          int r  = k - ci * KSQ;
          int ry = r / KS;
          iy = oy * stride + ry - (KS / 2);
          ix = ox * stride + (r - ry * KS) - (KS / 2);
        }
        if (iy >= 0 && iy < H && ix >= 0 && ix < W)
          v = x[(size_t)ci * H * W + (size_t)iy * W + ix];
      }
      b[j] = (_Float16)v;
    }

    // MT WMMAs reuse the same B fragment
#pragma unroll
    for (int i = 0; i < MT; ++i) {
      v16h a = *(const v16h*)(As + i * 512 + lane * 16);
      acc[i] = __builtin_amdgcn_wmma_f32_16x16x32_f16(
          false, a, false, b, (short)0, acc[i], false, false);
    }
  }

#pragma unroll
  for (int i = 0; i < MT; ++i) {
#pragma unroll
    for (int r = 0; r < 8; ++r) {
      int m = (mbase + i) * 16 + half * 8 + r;
      if (m < Cout && pv) {
        float v = acc[i][r] + bias[m];
        if (act == 1)      v = fmaxf(v, 0.f);
        else if (act == 2) v = v > 0.f ? v : 0.1f * v;
        y[(size_t)m * npix + pix] = v;
      }
    }
  }
}

// ---------------------------------------------------------------------------
// Bilinear sample, reference semantics (per-tap validity, zero outside).
// ---------------------------------------------------------------------------
__device__ __forceinline__ float bilin(const float* __restrict__ img, int H, int W,
                                       float ys, float xs) {
  float y0 = floorf(ys), x0 = floorf(xs);
  float wy = ys - y0, wx = xs - x0;
  float acc = 0.f;
#pragma unroll
  for (int dy = 0; dy < 2; ++dy) {
#pragma unroll
    for (int dx = 0; dx < 2; ++dx) {
      float yi = y0 + dy, xi = x0 + dx;
      float wgt = (dy ? wy : 1.f - wy) * (dx ? wx : 1.f - wx);
      if (yi >= 0.f && yi <= (float)(H - 1) && xi >= 0.f && xi <= (float)(W - 1))
        acc += img[(int)yi * W + (int)xi] * wgt;
    }
  }
  return acc;
}

__device__ __forceinline__ float sigm(float v) { return 1.f / (1.f + expf(-v)); }

// aligned[t] = (t==0) ? feat[0] : flow_warp(feat[t-1], mv[t-1])
__global__ void align_warp_kernel(const float* __restrict__ feat, const float* __restrict__ mv,
                                  float* __restrict__ aligned) {
  size_t idx = (size_t)blockIdx.x * 256 + threadIdx.x;
  size_t tot = (size_t)TT * NF * NPIX;
  if (idx >= tot) return;
  int pix = (int)(idx % NPIX);
  int c   = (int)((idx / NPIX) % NF);
  int t   = (int)(idx / ((size_t)NF * NPIX));
  float out;
  if (t == 0) {
    out = feat[idx];
  } else {
    const float* img = feat + ((size_t)(t - 1) * NF + c) * NPIX;
    const float* fl  = mv + ((size_t)(t - 1) * NPIX + pix) * 2;
    int y = pix / WW, x = pix % WW;
    out = bilin(img, HH, WW, (float)y + fl[1], (float)x + fl[0]);
  }
  aligned[idx] = out;
}

// a12 = aligned*sigmoid(aligned*ref) + pred*sigmoid(pred*ref), ref = feat[T/2]
__global__ void gate_kernel(const float* __restrict__ feat, const float* __restrict__ aligned,
                            const float* __restrict__ pred, float* __restrict__ a12) {
  size_t idx = (size_t)blockIdx.x * 256 + threadIdx.x;
  size_t tot = (size_t)TT * NF * NPIX;
  if (idx >= tot) return;
  int pix = (int)(idx % NPIX);
  int c   = (int)((idx / NPIX) % NF);
  float ref = feat[((size_t)(TT / 2) * NF + c) * NPIX + pix];
  float al = aligned[idx];
  float pr = pred[idx];
  a12[idx] = al * sigm(al * ref) + pr * sigm(pr * ref);
}

// out = 0.1*a + b
__global__ void axpy_kernel(const float* __restrict__ a, const float* __restrict__ b,
                            float* __restrict__ out, int n) {
  int idx = blockIdx.x * 256 + threadIdx.x;
  if (idx >= n) return;
  out[idx] = 0.1f * a[idx] + b[idx];
}

// Transposed conv 4x4 stride 2 pad 1 (+bias, relu). w is [Cin][Cout][4][4];
// effective tap: wf[o][c][ky][kx] = w[c][o][3-ky][3-kx].
__global__ void convT4x4_kernel(const float* __restrict__ in, const float* __restrict__ w,
                                const float* __restrict__ b, float* __restrict__ out,
                                int C, int Hi, int Wi) {
  int Ho = 2 * Hi, Wo = 2 * Wi;
  int npo = Ho * Wo;
  int idx = blockIdx.x * 256 + threadIdx.x;
  if (idx >= C * npo) return;
  int pix = idx % npo;
  int o   = idx / npo;
  int y = pix / Wo, x = pix % Wo;
  float acc = b[o];
  for (int ky = 0; ky < 4; ++ky) {
    int u = y + ky - 2;
    if (u < 0 || u >= 2 * Hi || (u & 1)) continue;
    int iy = u >> 1;
    for (int kx = 0; kx < 4; ++kx) {
      int v = x + kx - 2;
      if (v < 0 || v >= 2 * Wi || (v & 1)) continue;
      int ix = v >> 1;
      const float* ip = in + (size_t)iy * Wi + ix;
      const float* wp = w + (size_t)o * 16 + (3 - ky) * 4 + (3 - kx);
      for (int c = 0; c < C; ++c)
        acc += ip[(size_t)c * Hi * Wi] * wp[(size_t)c * C * 16];
    }
  }
  out[idx] = acc > 0.f ? acc : 0.f;
}

// Deformable sampling + modulation -> mod[(g*9+k)][pix]  (63 planes)
// om layout: [0:126) offsets (g*18 + k*2 + {y,x}), [126:189) masks.
__global__ void deform_mod_kernel(const float* __restrict__ lqs, const float* __restrict__ om,
                                  float* __restrict__ mod) {
  size_t idx = (size_t)blockIdx.x * 256 + threadIdx.x;
  size_t tot = (size_t)63 * NPIX;
  if (idx >= tot) return;
  int pix = (int)(idx % NPIX);
  int gk  = (int)(idx / NPIX);
  int g = gk / 9, kk = gk % 9;
  int y = pix / WW, x = pix % WW;
  float offy = om[((size_t)(g * 18 + kk * 2 + 0)) * NPIX + pix];
  float offx = om[((size_t)(g * 18 + kk * 2 + 1)) * NPIX + pix];
  float m    = sigm(om[((size_t)(126 + gk)) * NPIX + pix]);
  float sy = (float)(y + kk / 3 - 1) + offy;
  float sx = (float)(x + kk % 3 - 1) + offx;
  mod[idx] = bilin(lqs + (size_t)g * NPIX, HH, WW, sy, sx) * m;
}

// ---------------------------------------------------------------------------
// Host-side conv dispatch
// ---------------------------------------------------------------------------
static void run_conv(const float* x, const _Float16* Ap, const float* bias, float* y,
                     int Cin, int H, int W, int Cout, int stride, int ks, int act,
                     int frames, hipStream_t stream) {
  int Ho = (stride == 1) ? H : H / 2;
  int Wo = (stride == 1) ? W : W / 2;
  int ktiles = ceil_div_h(Cin * ks * ks, 32);
  int mt = ceil_div_h(ceil_div_h(Cout, 16), MT_TILES);  // blocks of MT tiles
  dim3 g(mt, ceil_div_h(Ho * Wo, 64), frames);
  if (ks == 3)
    conv_wmma_kernel<3, MT_TILES><<<g, 128, 0, stream>>>(
        x, Ap, bias, y, Cin, H, W, Cout, Ho, Wo, stride, ktiles, act);
  else
    conv_wmma_kernel<1, MT_TILES><<<g, 128, 0, stream>>>(
        x, Ap, bias, y, Cin, H, W, Cout, Ho, Wo, stride, ktiles, act);
}

// ---------------------------------------------------------------------------
// Host orchestration
// ---------------------------------------------------------------------------
extern "C" void kernel_launch(void* const* d_in, const int* in_sizes, int n_in,
                              void* d_out, int out_size, void* d_ws, size_t ws_size,
                              hipStream_t stream) {
  (void)in_sizes; (void)n_in; (void)out_size; (void)ws_size;

  const float* lqs    = (const float*)d_in[0];
  const float* preds  = (const float*)d_in[1];
  const float* mv     = (const float*)d_in[2];
  const float* bw1    = (const float*)d_in[3];
  const float* bb1    = (const float*)d_in[4];
  const float* bw2    = (const float*)d_in[5];
  const float* bb2    = (const float*)d_in[6];
  const float* dn1_w  = (const float*)d_in[7];
  const float* dn1_b  = (const float*)d_in[8];
  const float* dn2_w  = (const float*)d_in[9];
  const float* dn2_b  = (const float*)d_in[10];
  const float* up1_cw = (const float*)d_in[11];
  const float* up1_cb = (const float*)d_in[12];
  const float* up1_tw = (const float*)d_in[13];
  const float* up1_tb = (const float*)d_in[14];
  const float* up2_cw = (const float*)d_in[15];
  const float* up2_cb = (const float*)d_in[16];
  const float* up2_tw = (const float*)d_in[17];
  const float* up2_tb = (const float*)d_in[18];
  const float* tr_cw  = (const float*)d_in[19];
  const float* tr_cb  = (const float*)d_in[20];
  const float* tr_tw  = (const float*)d_in[21];
  const float* tr_tb  = (const float*)d_in[22];
  const float* ff_w   = (const float*)d_in[23];
  const float* ff_b   = (const float*)d_in[24];
  const float* om_w   = (const float*)d_in[25];
  const float* om_b   = (const float*)d_in[26];
  const float* dc_w   = (const float*)d_in[27];
  const float* dc_b   = (const float*)d_in[28];

  // bump allocator on workspace
  char* ws = (char*)d_ws;
  size_t off = 0;
  auto alloc = [&](size_t bytes) -> void* {
    void* r = ws + off;
    off += (bytes + 255) & ~(size_t)255;
    return r;
  };
  auto allocF = [&](size_t elems) -> float* { return (float*)alloc(elems * 4); };

  // f32 intermediates
  float* tmp1   = allocF((size_t)TT * NF * NPIX);  // bb stage / reused as a12
  float* feat   = allocF((size_t)TT * NF * NPIX);
  float* predf  = allocF((size_t)TT * NF * NPIX);
  float* aligned= allocF((size_t)TT * NF * NPIX);
  float* fuseA  = allocF((size_t)NF * NPIX);
  float* fuseB  = allocF((size_t)NF * NPIX);
  float* out0   = allocF((size_t)NF * NPIX);
  float* out1   = allocF((size_t)NF * 96 * 96);
  float* out2   = allocF((size_t)NF * 48 * 48);
  float* t0     = allocF((size_t)NF * 24 * 24);
  float* t1     = allocF((size_t)NF * 48 * 48);
  float* cat2   = allocF((size_t)2 * NF * 48 * 48);
  float* u2c    = allocF((size_t)NF * 48 * 48);
  float* u2t    = allocF((size_t)NF * 96 * 96);
  float* cat1   = allocF((size_t)2 * NF * 96 * 96);
  float* u1c    = allocF((size_t)NF * 96 * 96);
  float* featf  = allocF((size_t)NF * NPIX);
  float* offmsk = allocF((size_t)189 * NPIX);
  float* modb   = allocF((size_t)63 * NPIX);

  // packed f16 weights (mt padded to multiple of MT_TILES, zero-filled)
  auto pack = [&](const float* w, int Cout, int Cin, int ks) -> _Float16* {
    int Ktot = Cin * ks * ks;
    int mt = ceil_div_h(ceil_div_h(Cout, 16), MT_TILES) * MT_TILES;
    int kt = ceil_div_h(Ktot, 32);
    size_t n = (size_t)mt * kt * 512;
    _Float16* Ap = (_Float16*)alloc(n * sizeof(_Float16));
    pack_w_kernel<<<ceil_div_h((int)n, 256), 256, 0, stream>>>(w, Ap, Cout, Ktot, mt, kt);
    return Ap;
  };
  _Float16* A_bw1  = pack(bw1,  NF, 1,   3);
  _Float16* A_bw2  = pack(bw2,  NF, NF,  3);
  _Float16* A_dn1  = pack(dn1_w,NF, NF,  3);
  _Float16* A_dn2  = pack(dn2_w,NF, NF,  3);
  _Float16* A_trc  = pack(tr_cw,NF, NF,  3);
  _Float16* A_u2c  = pack(up2_cw,NF, 2*NF, 3);
  _Float16* A_u1c  = pack(up1_cw,NF, 2*NF, 3);
  _Float16* A_ff   = pack(ff_w, NF, TT*NF, 3);
  _Float16* A_om   = pack(om_w, TT*3*9, NF, 3);
  _Float16* A_dc   = pack(dc_w, NF, TT*9, 1);   // 1x1 GEMM over 63 deform planes

  // --- backbone on lqs and preds (no activation) ---
  run_conv(lqs,   A_bw1, bb1, tmp1, 1,  HH, WW, NF, 1, 3, 0, TT, stream);
  run_conv(tmp1,  A_bw2, bb2, feat, NF, HH, WW, NF, 1, 3, 0, TT, stream);
  run_conv(preds, A_bw1, bb1, tmp1, 1,  HH, WW, NF, 1, 3, 0, TT, stream);
  run_conv(tmp1,  A_bw2, bb2, predf,NF, HH, WW, NF, 1, 3, 0, TT, stream);

  // --- alignment (flow warp) ---
  {
    size_t tot = (size_t)TT * NF * NPIX;
    align_warp_kernel<<<(int)((tot + 255) / 256), 256, 0, stream>>>(feat, mv, aligned);
  }

  // --- fuse path: fuseB from raw feats first, then gate into a12 (= tmp1) ---
  run_conv(feat, A_ff, ff_b, fuseB, TT * NF, HH, WW, NF, 1, 3, 2, 1, stream);  // lrelu
  {
    size_t tot = (size_t)TT * NF * NPIX;
    gate_kernel<<<(int)((tot + 255) / 256), 256, 0, stream>>>(feat, aligned, predf, tmp1);
  }
  run_conv(tmp1, A_ff, ff_b, fuseA, TT * NF, HH, WW, NF, 1, 3, 2, 1, stream);  // lrelu
  axpy_kernel<<<ceil_div_h(NF * NPIX, 256), 256, 0, stream>>>(fuseA, fuseB, out0, NF * NPIX);

  // --- U-Net ---
  run_conv(out0, A_dn1, dn1_b, out1, NF, HH, WW, NF, 2, 3, 1, 1, stream);  // 96x96
  run_conv(out1, A_dn2, dn2_b, out2, NF, 96, 96, NF, 2, 3, 1, 1, stream);  // 48x48
  run_conv(out2, A_trc, tr_cb, t0,   NF, 48, 48, NF, 2, 3, 1, 1, stream);  // 24x24
  convT4x4_kernel<<<ceil_div_h(NF * 48 * 48, 256), 256, 0, stream>>>(t0, tr_tw, tr_tb, t1, NF, 24, 24);

  hipMemcpyAsync(cat2, t1, (size_t)NF * 48 * 48 * 4, hipMemcpyDeviceToDevice, stream);
  hipMemcpyAsync(cat2 + (size_t)NF * 48 * 48, out2, (size_t)NF * 48 * 48 * 4,
                 hipMemcpyDeviceToDevice, stream);
  run_conv(cat2, A_u2c, up2_cb, u2c, 2 * NF, 48, 48, NF, 1, 3, 1, 1, stream);
  convT4x4_kernel<<<ceil_div_h(NF * 96 * 96, 256), 256, 0, stream>>>(u2c, up2_tw, up2_tb, u2t, NF, 48, 48);

  hipMemcpyAsync(cat1, u2t, (size_t)NF * 96 * 96 * 4, hipMemcpyDeviceToDevice, stream);
  hipMemcpyAsync(cat1 + (size_t)NF * 96 * 96, out1, (size_t)NF * 96 * 96 * 4,
                 hipMemcpyDeviceToDevice, stream);
  run_conv(cat1, A_u1c, up1_cb, u1c, 2 * NF, 96, 96, NF, 1, 3, 1, 1, stream);
  convT4x4_kernel<<<ceil_div_h(NF * NPIX, 256), 256, 0, stream>>>(u1c, up1_tw, up1_tb, featf, NF, 96, 96);

  // --- offsets/masks + deformable conv ---
  run_conv(featf, A_om, om_b, offmsk, NF, HH, WW, TT * 3 * 9, 1, 3, 0, 1, stream);  // 189 ch
  {
    size_t tot = (size_t)63 * NPIX;
    deform_mod_kernel<<<(int)((tot + 255) / 256), 256, 0, stream>>>(lqs, offmsk, modb);
  }
  // final 64x63 GEMM over pixels (1x1 conv) + bias + relu -> d_out
  run_conv(modb, A_dc, dc_b, (float*)d_out, TT * 9, HH, WW, NF, 1, 1, 1, 1, stream);
}
